// SimpleJoiner_12249246729036
// MI455X (gfx1250) — matrix operand order repaired and checked by reading, not compiled, not run on hardware
//
#include <hip/hip_runtime.h>

#define B_   8
#define T_   256
#define U_   64
#define ED   512      // ENC_DIM == DEC_DIM
#define JD   640      // JOINT_DIM
#define VOC  1024     // VOCAB

typedef __attribute__((ext_vector_type(16))) _Float16 v16h;
typedef __attribute__((ext_vector_type(8)))  _Float16 v8h;
typedef __attribute__((ext_vector_type(4)))  _Float16 v4h;
typedef __attribute__((ext_vector_type(8)))  float    v8f;

// ---------------------------------------------------------------------------
// Stage 1: Y[row, j] = dot(X[row, :], W[j, :])   (x @ W.T), f32 throughout.
// One block per row; x row staged in LDS; each thread owns one of 640 outputs.
// ---------------------------------------------------------------------------
__global__ __launch_bounds__(JD) void proj_kernel(const float* __restrict__ X,
                                                  const float* __restrict__ W,
                                                  float* __restrict__ Y) {
    __shared__ float sx[ED];
    const int row = blockIdx.x;
    const int tid = threadIdx.x;
    if (tid < ED) sx[tid] = X[(size_t)row * ED + tid];
    __syncthreads();

    const float4* wp  = (const float4*)(W + (size_t)tid * ED);
    const float4* sx4 = (const float4*)sx;
    float acc = 0.f;
    #pragma unroll 4
    for (int k = 0; k < ED / 4; ++k) {
        float4 w = wp[k];
        float4 x = sx4[k];
        acc += w.x * x.x + w.y * x.y + w.z * x.z + w.w * x.w;
    }
    Y[(size_t)row * JD + tid] = acc;
}

// ---------------------------------------------------------------------------
// Stage 2: W_out f32 -> f16 (row-major [VOC, JD]) so the B operand of WMMA
// can be loaded contiguously per lane.
// ---------------------------------------------------------------------------
__global__ __launch_bounds__(256) void cvt16_kernel(const float* __restrict__ src,
                                                    _Float16* __restrict__ dst,
                                                    int n4) {
    int i = blockIdx.x * blockDim.x + threadIdx.x;
    if (i < n4) {
        float4 v = ((const float4*)src)[i];
        v4h h = { (_Float16)v.x, (_Float16)v.y, (_Float16)v.z, (_Float16)v.w };
        ((v4h*)dst)[i] = h;
    }
}

// ---------------------------------------------------------------------------
// Stage 3: joint = silu(enc_p[b,t,:] + pred_p[b,u,:]); out = joint @ W_out.T
//
// One workgroup (256 threads, 8 waves) per (b,t): all 64 u-rows x 1024 cols.
// Joint tile staged as f16 in 80 KB dynamic LDS. Each wave owns a disjoint
// 128-column strip (so every W_out element is fetched from L2 exactly once
// per workgroup) and reuses each B fragment across 4 register-resident
// M-tiles: per k-step, 2 B tiles x 4 M tiles = 8 v_wmma_f32_16x16x32_f16.
// K-loop is kept rolled (#pragma unroll 1) so the live set stays ~140 VGPRs
// and the compiler does not spill fragments to scratch.
// ---------------------------------------------------------------------------
__global__ __launch_bounds__(256) void joint_gemm_kernel(
        const float* __restrict__ enc_p,   // [B*T, JD]
        const float* __restrict__ pred_p,  // [B*U, JD]
        const _Float16* __restrict__ w16,  // [VOC, JD] row-major
        float* __restrict__ out)           // [B*T*U, VOC]
{
    extern __shared__ _Float16 sJ[];       // 64 * JD halves = 81920 bytes

    const int tid = threadIdx.x;
    const int bt  = blockIdx.x;
    const int b   = bt / T_;

    // ---- stage silu(enc + pred) for all 64 u rows into LDS as f16 ----------
    const float* er = enc_p + (size_t)bt * JD;
    const float* pr = pred_p + (size_t)(b * U_) * JD;
    for (int i = tid; i < U_ * JD; i += 256) {   // 40960 elems, uniform count
        int r = i / JD;
        int c = i - r * JD;
        float x = er[c] + pr[(size_t)r * JD + c];
        float s = x / (1.f + __expf(-x));        // SiLU
        sJ[i] = (_Float16)s;
    }
    __syncthreads();

    const int lane = tid & 31;
    const int wave = tid >> 5;                   // 0..7 -> column strip
    const int hi   = (lane >> 4) & 1;            // lanes 16..31?
    const int lrow = lane & 15;

    // A-fragment (16-bit A 16x32): lanes 0-15 row=lane, K {0..15};
    //                              lanes 16-31 row=lane-16, K {16..31}
    const int aOfs = hi ? 16 : 0;
    // B-fragment (16-bit B 32x16): lane%16 = column; lanes 16-31 take K+16.
    const int vcol  = lrow;
    const int kBofs = hi ? 16 : 0;
    // D layout: VGPR r -> (M = r + mofs, N = lane%16)
    const int mofs = hi ? 8 : 0;

    // Per-lane LDS base for A fragments (row lrow of m-tile 0)
    const _Float16* aBase = sJ + (size_t)lrow * JD + aOfs;

    float* outRow = out + (size_t)bt * U_ * VOC; // u-row stride = VOC

    #pragma unroll 1
    for (int np = 0; np < 4; ++np) {
        const int n0 = wave * 128 + np * 32;     // two 16-col tiles: n0, n0+16
        const _Float16* bBase0 = w16 + (size_t)(n0 +  0 + vcol) * JD + kBofs;
        const _Float16* bBase1 = w16 + (size_t)(n0 + 16 + vcol) * JD + kBofs;

        v8f acc[4][2];
        #pragma unroll
        for (int m = 0; m < 4; ++m) { acc[m][0] = (v8f){}; acc[m][1] = (v8f){}; }

        #pragma unroll 1
        for (int k0 = 0; k0 < JD; k0 += 32) {
            // B fragments: contiguous 32-byte per-lane loads from f16 W_out
            v16h Bf0 = __builtin_shufflevector(*(const v8h*)(bBase0 + k0),
                                               *(const v8h*)(bBase0 + k0 + 8),
                         0,1,2,3,4,5,6,7,8,9,10,11,12,13,14,15);
            v16h Bf1 = __builtin_shufflevector(*(const v8h*)(bBase1 + k0),
                                               *(const v8h*)(bBase1 + k0 + 8),
                         0,1,2,3,4,5,6,7,8,9,10,11,12,13,14,15);

            #pragma unroll
            for (int m = 0; m < 4; ++m) {
                const _Float16* ap = aBase + (size_t)(m * 16) * JD + k0;
                v16h A = __builtin_shufflevector(*(const v8h*)(ap),
                                                 *(const v8h*)(ap + 8),
                           0,1,2,3,4,5,6,7,8,9,10,11,12,13,14,15);

                acc[m][0] = __builtin_amdgcn_wmma_f32_16x16x32_f16(
                                false, A, false, Bf0, (short)0, acc[m][0], false, false);
                acc[m][1] = __builtin_amdgcn_wmma_f32_16x16x32_f16(
                                false, A, false, Bf1, (short)0, acc[m][1], false, false);
            }
        }

        // ---- store: D VGPR r holds (M = r + mofs, N = lane%16) -------------
        #pragma unroll
        for (int m = 0; m < 4; ++m) {
            #pragma unroll
            for (int r = 0; r < 8; ++r) {
                size_t ro = (size_t)(m * 16 + mofs + r) * VOC;
                outRow[ro + n0 +  0 + vcol] = acc[m][0][r];
                outRow[ro + n0 + 16 + vcol] = acc[m][1][r];
            }
        }
    }
}

// ---------------------------------------------------------------------------
extern "C" void kernel_launch(void* const* d_in, const int* in_sizes, int n_in,
                              void* d_out, int out_size, void* d_ws, size_t ws_size,
                              hipStream_t stream) {
    const float* enc  = (const float*)d_in[0];   // [B,T,ED]
    const float* pred = (const float*)d_in[1];   // [B,U,ED]
    const float* We   = (const float*)d_in[2];   // [JD,ED]
    const float* Wp   = (const float*)d_in[3];   // [JD,ED]
    const float* Wo   = (const float*)d_in[4];   // [VOC,JD]
    float* out = (float*)d_out;                  // [B,T,U,VOC]

    float*    enc_p  = (float*)d_ws;                               // B*T*JD f32
    float*    pred_p = enc_p + (size_t)B_ * T_ * JD;               // B*U*JD f32
    _Float16* w16    = (_Float16*)(pred_p + (size_t)B_ * U_ * JD); // VOC*JD f16

    proj_kernel<<<B_ * T_, JD, 0, stream>>>(enc,  We, enc_p);
    proj_kernel<<<B_ * U_, JD, 0, stream>>>(pred, Wp, pred_p);

    const int n4 = (VOC * JD) / 4;
    cvt16_kernel<<<(n4 + 255) / 256, 256, 0, stream>>>(Wo, w16, n4);

    // 80 KB dynamic LDS for the 64x640 f16 joint tile
    joint_gemm_kernel<<<B_ * T_, 256, U_ * JD * sizeof(_Float16), stream>>>(
        enc_p, pred_p, w16, out);
}
